// AttentionBranch_42554535969385
// MI455X (gfx1250) — matrix operand order repaired and checked by reading, not compiled, not run on hardware
//
#include <hip/hip_runtime.h>

#define NUM_HEADS 32
#define NUM_KV    8
#define HEAD_DIM  128
#define WINDOW    256
#define NUM_META  128
#define QTILE     16
#define KTILE     32

typedef __attribute__((ext_vector_type(16))) __bf16   bf16x16;
typedef __attribute__((ext_vector_type(8)))  float    f32x8;
typedef __attribute__((ext_vector_type(4)))  unsigned u32x4;
typedef __attribute__((ext_vector_type(4)))  float    f32x4;

union AFrag { unsigned u[8]; bf16x16 v; };
union BFrag { u32x4 q[2];    bf16x16 v; };

// LDS strides (bf16 elements): 16B-aligned rows + bank rotation for conflict-free b128
#define QSTRIDE 130
#define KSTRIDE 136
#define VSTRIDE 40
#define PSTRIDE 34

#define NLFREQ  (-0.14391156f)          // -ln(10000)/64
#define KSCALE  0.0883883476483184f     // 1/sqrt(128)

// ---------------- pass 1: RoPE K -> bf16 [s][kvh][d]; transpose V -> bf16 [kvh][d][s] ----
__global__ __launch_bounds__(256)
void prep_kv(const float* __restrict__ Kg, const float* __restrict__ Vg,
             __bf16* __restrict__ Kr, __bf16* __restrict__ Vt, int S)
{
  int idx = blockIdx.x * 256 + threadIdx.x;
  if (idx >= S * NUM_KV * HEAD_DIM) return;
  int d   = idx & 127;
  int kvh = (idx >> 7) & 7;
  int s   = idx >> 10;
  const float* kp = Kg + (size_t)s * (NUM_KV * HEAD_DIM) + kvh * HEAD_DIM;
  float x = kp[d], y = kp[d ^ 64];
  float ang = (float)s * __expf((float)(d & 63) * NLFREQ);
  float c = __cosf(ang), sn = __sinf(ang);
  float rk = (d < 64) ? (x * c - y * sn) : (x * c + y * sn);
  Kr[idx] = (__bf16)rk;
  Vt[((size_t)kvh * HEAD_DIM + d) * S + s] = (__bf16)Vg[idx];
}

// ---------------- pass 2: flash attention over preprocessed bf16 K/V ----------------
__global__ __launch_bounds__(128, 1)
void fa_fast(const float* __restrict__ Qg, const __bf16* __restrict__ Kr,
             const __bf16* __restrict__ Vt, float* __restrict__ Og, int S)
{
  __shared__ __bf16 sQ[4][QTILE * QSTRIDE];
  __shared__ __bf16 sK[KTILE * KSTRIDE];
  __shared__ __bf16 sVt[HEAD_DIM * VSTRIDE];
  __shared__ __bf16 sP[4][QTILE * PSTRIDE];

  const int tid  = threadIdx.x;
  const int lane = tid & 31;
  const int wv   = tid >> 5;
  const int hi   = lane >> 4;
  const int nn   = lane & 15;
  const int qt   = blockIdx.x;
  const int kvh  = blockIdx.y;
  const int h    = kvh * 4 + wv;
  const int q0   = qt * QTILE;
  const int qend = q0 + QTILE - 1;

  // ---- Q tile: float4 pair loads, RoPE (cos/sin shared by mirrored halves), scale ----
  for (int e = lane; e < QTILE * 16; e += 32) {
    int row = e >> 4, qd = (e & 15) * 4;          // qd in [0,60]
    int pos = q0 + row;
    int posc = (pos < S) ? pos : (S - 1);
    const float* qp = Qg + (size_t)posc * (NUM_HEADS * HEAD_DIM) + h * HEAD_DIM;
    f32x4 x = *(const f32x4*)&qp[qd];
    f32x4 y = *(const f32x4*)&qp[qd + 64];
    #pragma unroll
    for (int j = 0; j < 4; ++j) {
      float ang = (float)pos * __expf((float)(qd + j) * NLFREQ);
      float c = __cosf(ang), sn = __sinf(ang);
      sQ[wv][row * QSTRIDE + qd + j]      = (__bf16)((x[j] * c - y[j] * sn) * KSCALE);
      sQ[wv][row * QSTRIDE + qd + 64 + j] = (__bf16)((y[j] * c + x[j] * sn) * KSCALE);
    }
  }

  // ---- accumulators ----
  const f32x8 vzero = {0.f, 0.f, 0.f, 0.f, 0.f, 0.f, 0.f, 0.f};
  f32x8 acco[8];
  #pragma unroll
  for (int t = 0; t < 8; ++t) acco[t] = vzero;
  float m_i[8], l_i[8];
  #pragma unroll
  for (int r = 0; r < 8; ++r) { m_i[r] = -3.0e38f; l_i[r] = 0.0f; }

  const __bf16* Krh = Kr + (size_t)kvh * HEAD_DIM;            // + s*1024 + d
  const __bf16* Vth = Vt + (size_t)kvh * HEAD_DIM * S;        // + d*S + key
  const int klb = tid >> 4;        // K staging: base key row
  const int d8  = tid & 15;        // K staging: 8-elem chunk within row

  u32x4 kst[4], vst[4];            // staged next tile (software pipeline)

  auto stage_load = [&](int k0) {
    #pragma unroll
    for (int i = 0; i < 4; ++i) {
      int kr = k0 + klb + 8 * i;   kr = (kr < S) ? kr : (S - 1);
      kst[i] = *(const u32x4*)(Krh + (size_t)kr * (NUM_KV * HEAD_DIM) + d8 * 8);
      int kc = k0 + 8 * i;         kc = (kc + 8 <= S) ? kc : (S - 8);
      vst[i] = *(const u32x4*)(Vth + (size_t)tid * S + kc);
    }
  };
  auto stage_store = [&]() {
    #pragma unroll
    for (int i = 0; i < 4; ++i) {
      *(u32x4*)&sK[(klb + 8 * i) * KSTRIDE + d8 * 8] = kst[i];
      *(u32x4*)&sVt[tid * VSTRIDE + 8 * i] = vst[i];
    }
  };

  int k0 = 0;
  stage_load(0);                   // tile 0 always live (meta prefix)
  while (true) {
    __syncthreads();               // previous tile's LDS reads done
    stage_store();

    // next live tile (skip fully-masked gap between meta prefix and window)
    int k1 = k0 + KTILE;
    if (k1 >= NUM_META && (k1 + KTILE - 1) < (q0 - WINDOW))
      k1 = ((q0 - WINDOW) >> 5) << 5;
    const bool have_next = (k1 <= qend);
    if (have_next) stage_load(k1); // prefetch overlaps compute below
    __syncthreads();               // this tile's LDS writes visible

    // ---- scores: two 16x16 sub-tiles over this 32-key block ----
    f32x8 sc[2];
    sc[0] = vzero; sc[1] = vzero;
    #pragma unroll
    for (int kk = 0; kk < 4; ++kk) {
      AFrag a;
      #pragma unroll
      for (int j = 0; j < 8; ++j) {
        int kb = 32 * kk + 2 * j + ((j >= 4) ? 8 : 0) + 8 * hi;
        a.u[j] = *(const unsigned*)&sQ[wv][nn * QSTRIDE + kb];
      }
      #pragma unroll
      for (int st = 0; st < 2; ++st) {
        BFrag b;
        const __bf16* bp = &sK[(st * 16 + nn) * KSTRIDE + 32 * kk + 8 * hi];
        b.q[0] = *(const u32x4*)bp;
        b.q[1] = *(const u32x4*)(bp + 16);
        sc[st] = __builtin_amdgcn_wmma_f32_16x16x32_bf16(
            false, a.v, false, b.v, (short)0, sc[st], false, false);
      }
    }

    // ---- mask + online softmax ----
    float p0[8], p1[8], scl[8];
    #pragma unroll
    for (int r = 0; r < 8; ++r) {
      int qrow = q0 + r + 8 * hi;
      int key0 = k0 + nn, key1 = key0 + 16;
      bool ok0 = (key0 <= qrow) && ((qrow - key0 <= WINDOW) || (key0 < NUM_META));
      bool ok1 = (key1 <= qrow) && ((qrow - key1 <= WINDOW) || (key1 < NUM_META));
      float s0 = ok0 ? sc[0][r] : -3.0e38f;
      float s1 = ok1 ? sc[1][r] : -3.0e38f;
      float rm = fmaxf(s0, s1);
      rm = fmaxf(rm, __shfl_xor(rm, 1));
      rm = fmaxf(rm, __shfl_xor(rm, 2));
      rm = fmaxf(rm, __shfl_xor(rm, 4));
      rm = fmaxf(rm, __shfl_xor(rm, 8));
      float mnew = fmaxf(m_i[r], rm);
      scl[r] = __expf(m_i[r] - mnew);
      p0[r]  = __expf(s0 - mnew);
      p1[r]  = __expf(s1 - mnew);
      float rs = p0[r] + p1[r];
      rs += __shfl_xor(rs, 1);
      rs += __shfl_xor(rs, 2);
      rs += __shfl_xor(rs, 4);
      rs += __shfl_xor(rs, 8);
      l_i[r] = l_i[r] * scl[r] + rs;
      m_i[r] = mnew;
    }
    #pragma unroll
    for (int t = 0; t < 8; ++t) {
      #pragma unroll
      for (int r = 0; r < 8; ++r) acco[t][r] *= scl[r];
    }

    // ---- P (D-layout) -> LDS -> A-layout ----
    #pragma unroll
    for (int r = 0; r < 8; ++r) {
      sP[wv][(r + 8 * hi) * PSTRIDE + nn]      = (__bf16)p0[r];
      sP[wv][(r + 8 * hi) * PSTRIDE + nn + 16] = (__bf16)p1[r];
    }
    __syncthreads();

    AFrag pa;
    #pragma unroll
    for (int j = 0; j < 8; ++j) {
      int kb = 2 * j + ((j >= 4) ? 8 : 0) + 8 * hi;
      pa.u[j] = *(const unsigned*)&sP[wv][nn * PSTRIDE + kb];
    }
    #pragma unroll
    for (int t = 0; t < 8; ++t) {
      BFrag b;
      const __bf16* bp = &sVt[(16 * t + nn) * VSTRIDE + 8 * hi];
      b.q[0] = *(const u32x4*)bp;
      b.q[1] = *(const u32x4*)(bp + 16);
      acco[t] = __builtin_amdgcn_wmma_f32_16x16x32_bf16(
          false, pa.v, false, b.v, (short)0, acco[t], false, false);
    }

    if (!have_next) break;
    k0 = k1;
  }

  // ---- normalize + store ----
  float rl[8];
  #pragma unroll
  for (int r = 0; r < 8; ++r) rl[r] = 1.0f / l_i[r];
  #pragma unroll
  for (int t = 0; t < 8; ++t) {
    #pragma unroll
    for (int r = 0; r < 8; ++r) {
      int row = q0 + r + 8 * hi;
      if (row < S) {
        Og[(size_t)row * (NUM_HEADS * HEAD_DIM) + h * HEAD_DIM + 16 * t + nn] =
            acco[t][r] * rl[r];
      }
    }
  }
}

// ---------------- fallback: fully self-contained (no workspace) ----------------
__global__ __launch_bounds__(128, 1)
void fa_swa_rope_kernel(const float* __restrict__ Qg, const float* __restrict__ Kg,
                        const float* __restrict__ Vg, float* __restrict__ Og, int S)
{
  __shared__ __bf16 sQ[4][QTILE * QSTRIDE];
  __shared__ __bf16 sK[KTILE * KSTRIDE];
  __shared__ __bf16 sVt[HEAD_DIM * VSTRIDE];
  __shared__ __bf16 sP[4][QTILE * PSTRIDE];

  const int tid = threadIdx.x, lane = tid & 31, wv = tid >> 5;
  const int hi = lane >> 4, nn = lane & 15;
  const int qt = blockIdx.x, kvh = blockIdx.y;
  const int h = kvh * 4 + wv;
  const int q0 = qt * QTILE, qend = q0 + QTILE - 1;

  for (int e = lane; e < QTILE * HEAD_DIM; e += 32) {
    int row = e >> 7, d = e & 127;
    int pos = q0 + row;
    float r = 0.0f;
    if (pos < S) {
      const float* qp = Qg + (size_t)pos * (NUM_HEADS * HEAD_DIM) + h * HEAD_DIM;
      float x = qp[d], y = qp[d ^ 64];
      float ang = (float)pos * __expf((float)(d & 63) * NLFREQ);
      float c = __cosf(ang), s = __sinf(ang);
      r = ((d < 64) ? (x * c - y * s) : (x * c + y * s)) * KSCALE;
    }
    sQ[wv][row * QSTRIDE + d] = (__bf16)r;
  }

  const f32x8 vzero = {0.f, 0.f, 0.f, 0.f, 0.f, 0.f, 0.f, 0.f};
  f32x8 acco[8];
  #pragma unroll
  for (int t = 0; t < 8; ++t) acco[t] = vzero;
  float m_i[8], l_i[8];
  #pragma unroll
  for (int r = 0; r < 8; ++r) { m_i[r] = -3.0e38f; l_i[r] = 0.0f; }

  for (int k0 = 0; k0 <= qend; k0 += KTILE) {
    if (k0 >= NUM_META && (k0 + KTILE - 1) < (q0 - WINDOW)) continue;
    __syncthreads();
    for (int e = tid; e < KTILE * HEAD_DIM; e += 128) {
      int kl = e >> 7, d = e & 127;
      int key = k0 + kl;
      float rk = 0.0f, vv = 0.0f;
      if (key < S) {
        const float* kp = Kg + (size_t)key * (NUM_KV * HEAD_DIM) + kvh * HEAD_DIM;
        float x = kp[d], y = kp[d ^ 64];
        float ang = (float)key * __expf((float)(d & 63) * NLFREQ);
        float c = __cosf(ang), s = __sinf(ang);
        rk = (d < 64) ? (x * c - y * s) : (x * c + y * s);
        vv = Vg[(size_t)key * (NUM_KV * HEAD_DIM) + kvh * HEAD_DIM + d];
      }
      sK[kl * KSTRIDE + d]  = (__bf16)rk;
      sVt[d * VSTRIDE + kl] = (__bf16)vv;
    }
    __syncthreads();

    f32x8 sc[2];
    sc[0] = vzero; sc[1] = vzero;
    #pragma unroll
    for (int kk = 0; kk < 4; ++kk) {
      AFrag a;
      #pragma unroll
      for (int j = 0; j < 8; ++j) {
        int kb = 32 * kk + 2 * j + ((j >= 4) ? 8 : 0) + 8 * hi;
        a.u[j] = *(const unsigned*)&sQ[wv][nn * QSTRIDE + kb];
      }
      #pragma unroll
      for (int st = 0; st < 2; ++st) {
        BFrag b;
        const __bf16* bp = &sK[(st * 16 + nn) * KSTRIDE + 32 * kk + 8 * hi];
        b.q[0] = *(const u32x4*)bp;
        b.q[1] = *(const u32x4*)(bp + 16);
        sc[st] = __builtin_amdgcn_wmma_f32_16x16x32_bf16(
            false, a.v, false, b.v, (short)0, sc[st], false, false);
      }
    }

    float p0[8], p1[8], scl[8];
    #pragma unroll
    for (int r = 0; r < 8; ++r) {
      int qrow = q0 + r + 8 * hi;
      int key0 = k0 + nn, key1 = key0 + 16;
      bool ok0 = (key0 <= qrow) && ((qrow - key0 <= WINDOW) || (key0 < NUM_META));
      bool ok1 = (key1 <= qrow) && ((qrow - key1 <= WINDOW) || (key1 < NUM_META));
      float s0 = ok0 ? sc[0][r] : -3.0e38f;
      float s1 = ok1 ? sc[1][r] : -3.0e38f;
      float rm = fmaxf(s0, s1);
      rm = fmaxf(rm, __shfl_xor(rm, 1));
      rm = fmaxf(rm, __shfl_xor(rm, 2));
      rm = fmaxf(rm, __shfl_xor(rm, 4));
      rm = fmaxf(rm, __shfl_xor(rm, 8));
      float mnew = fmaxf(m_i[r], rm);
      scl[r] = __expf(m_i[r] - mnew);
      p0[r]  = __expf(s0 - mnew);
      p1[r]  = __expf(s1 - mnew);
      float rs = p0[r] + p1[r];
      rs += __shfl_xor(rs, 1);
      rs += __shfl_xor(rs, 2);
      rs += __shfl_xor(rs, 4);
      rs += __shfl_xor(rs, 8);
      l_i[r] = l_i[r] * scl[r] + rs;
      m_i[r] = mnew;
    }
    #pragma unroll
    for (int t = 0; t < 8; ++t) {
      #pragma unroll
      for (int r = 0; r < 8; ++r) acco[t][r] *= scl[r];
    }

    #pragma unroll
    for (int r = 0; r < 8; ++r) {
      sP[wv][(r + 8 * hi) * PSTRIDE + nn]      = (__bf16)p0[r];
      sP[wv][(r + 8 * hi) * PSTRIDE + nn + 16] = (__bf16)p1[r];
    }
    __syncthreads();

    AFrag pa;
    #pragma unroll
    for (int j = 0; j < 8; ++j) {
      int kb = 2 * j + ((j >= 4) ? 8 : 0) + 8 * hi;
      pa.u[j] = *(const unsigned*)&sP[wv][nn * PSTRIDE + kb];
    }
    #pragma unroll
    for (int t = 0; t < 8; ++t) {
      BFrag b;
      const __bf16* bp = &sVt[(16 * t + nn) * VSTRIDE + 8 * hi];
      b.q[0] = *(const u32x4*)bp;
      b.q[1] = *(const u32x4*)(bp + 16);
      acco[t] = __builtin_amdgcn_wmma_f32_16x16x32_bf16(
          false, pa.v, false, b.v, (short)0, acco[t], false, false);
    }
  }

  float rl[8];
  #pragma unroll
  for (int r = 0; r < 8; ++r) rl[r] = 1.0f / l_i[r];
  #pragma unroll
  for (int t = 0; t < 8; ++t) {
    #pragma unroll
    for (int r = 0; r < 8; ++r) {
      int row = q0 + r + 8 * hi;
      if (row < S) {
        Og[(size_t)row * (NUM_HEADS * HEAD_DIM) + h * HEAD_DIM + 16 * t + nn] =
            acco[t][r] * rl[r];
      }
    }
  }
}

extern "C" void kernel_launch(void* const* d_in, const int* in_sizes, int n_in,
                              void* d_out, int out_size, void* d_ws, size_t ws_size,
                              hipStream_t stream) {
  (void)n_in; (void)out_size;
  const float* Q = (const float*)d_in[0];
  const float* K = (const float*)d_in[1];
  const float* V = (const float*)d_in[2];
  float* O = (float*)d_out;
  int S = in_sizes[0] / (NUM_HEADS * HEAD_DIM);   // B=1
  dim3 grid((S + QTILE - 1) / QTILE, NUM_KV, 1);

  size_t kvElems = (size_t)S * NUM_KV * HEAD_DIM;
  size_t need = kvElems * 2 /*bf16*/ * 2 /*K + V*/;
  if (d_ws && ws_size >= need) {
    __bf16* Kr = (__bf16*)d_ws;
    __bf16* Vt = Kr + kvElems;
    int total = (int)kvElems;
    prep_kv<<<(total + 255) / 256, 256, 0, stream>>>(K, V, Kr, Vt, S);
    fa_fast<<<grid, 128, 0, stream>>>(Q, Kr, Vt, O, S);
  } else {
    fa_swa_rope_kernel<<<grid, 128, 0, stream>>>(Q, K, V, O, S);
  }
}
